// M3GNet_60095182406217
// MI455X (gfx1250) — matrix operand order repaired and checked by reading, not compiled, not run on hardware
//
#include <hip/hip_runtime.h>

// ---------------------------------------------------------------------------
// M3GNet forward for MI455X (gfx1250, wave32, WMMA).
// Gated-MLP GEMMs run on v_wmma_f32_16x16x32_bf16 (f32 accumulate) with the
// full per-conv weight set staged in LDS (320KB/WGP) and 8 edge-tiles per
// workgroup, so B-fragments stream from LDS instead of thrashing L2.
// K=9 contractions stay on VALU (memory bound). Scatters use f32 atomics.
// ---------------------------------------------------------------------------

typedef __bf16 bf16_t;
typedef __attribute__((ext_vector_type(16))) __bf16 v16bf;
typedef __attribute__((ext_vector_type(8)))  float  v8f;

#define DFEAT 64
#define DEG   9
#define NBLK  3
#define NGRAPH 128

#define GWAVES 8                       // edge tiles (waves) per workgroup
#define PER_CONV_ELEMS 40960           // bf16 elems: W0(12288) W1 W2 V0(12288) V1 V2
#define SMEM_W_BYTES  (PER_CONV_ELEMS * 2)                 // 81920
#define SMEM_SRBF_OFF SMEM_W_BYTES                         // 81920
#define SMEM_SH_OFF   (SMEM_SRBF_OFF + GWAVES * 16 * 9 * 4)  // 86528 (16B aligned)
#define SMEM_TOTAL    (SMEM_SH_OFF + GWAVES * 16 * DFEAT * 2) // 102912

__device__ __forceinline__ float siluf(float x) { return x * (1.0f / (1.0f + __expf(-x))); }
__device__ __forceinline__ float sigmf_(float x) { return 1.0f / (1.0f + __expf(-x)); }

// closed-form spherical Bessel j_l, l = 0..3
__device__ __forceinline__ float sphj(int l, float x) {
  float s, c;
  __sincosf(x, &s, &c);
  float inv = 1.0f / x, inv2 = inv * inv;
  if (l == 0) return s * inv;
  if (l == 1) return s * inv2 - c * inv;
  if (l == 2) return (3.0f * inv2 * inv - inv) * s - 3.0f * inv2 * c;
  return (15.0f * inv2 * inv2 - 6.0f * inv2) * s - (15.0f * inv2 * inv - inv) * c;
}

// SphericalBesselFunction expansion -> 9 features (l-major)
__device__ __forceinline__ void sbf9(float r, float* out) {
  const float roots[3][3] = {
      {3.141592653589793f, 6.283185307179586f, 9.42477796076938f},
      {4.493409457909064f, 7.725251836937707f, 10.904121659428899f},
      {5.763459196894550f, 9.095011330476355f, 12.322940970566582f}};
  const float factor = 0.12649110640673517f;  // sqrt(2/5^3)
#pragma unroll
  for (int l = 0; l < 3; ++l) {
#pragma unroll
    for (int n = 0; n < 3; ++n) {
      float z = roots[l][n];
      float x = r * z * 0.2f;  // /CUTOFF(=5)
      float norm = fabsf(sphj(l + 1, z));
      out[l * 3 + n] = factor * sphj(l, x) / norm;
    }
  }
}

// ---------------------------------------------------------------------------
// 1) Per-edge geometry: bond_vec, dist, rbf[9], three-body cutoff
// ---------------------------------------------------------------------------
__global__ void k_edge_geom(const float* __restrict__ pos, const int* __restrict__ esrc,
                            const int* __restrict__ edst, float* __restrict__ bond_vec,
                            float* __restrict__ dist, float* __restrict__ rbf,
                            float* __restrict__ tbcut, int E) {
  int e = blockIdx.x * blockDim.x + threadIdx.x;
  if (e >= E) return;
  int s = esrc[e], d = edst[e];
  float vx = pos[d * 3 + 0] - pos[s * 3 + 0];
  float vy = pos[d * 3 + 1] - pos[s * 3 + 1];
  float vz = pos[d * 3 + 2] - pos[s * 3 + 2];
  float r = fmaxf(sqrtf(vx * vx + vy * vy + vz * vz), 1e-6f);
  bond_vec[e * 3 + 0] = vx;
  bond_vec[e * 3 + 1] = vy;
  bond_vec[e * 3 + 2] = vz;
  dist[e] = r;
  float rb[9];
  sbf9(r, rb);
#pragma unroll
  for (int j = 0; j < 9; ++j) rbf[(size_t)e * 9 + j] = rb[j];
  float x = r * 0.25f;  // /TB_CUTOFF(=4)
  float x2 = x * x;
  float f = 1.0f - 6.0f * x2 * x2 * x + 15.0f * x2 * x2 - 10.0f * x2 * x;
  tbcut[e] = (r <= 4.0f) ? f : 0.0f;
}

// ---------------------------------------------------------------------------
// 2) Node embedding gather, edge embedding MLP (K=9, VALU)
// ---------------------------------------------------------------------------
__global__ void k_node_init(const float* __restrict__ emb, const int* __restrict__ ntype,
                            float* __restrict__ node_feat, int N) {
  int idx = blockIdx.x * blockDim.x + threadIdx.x;
  if (idx >= N * DFEAT) return;
  int n = idx >> 6, d = idx & 63;
  node_feat[idx] = emb[ntype[n] * DFEAT + d];
}

__global__ void k_edge_init(const float* __restrict__ rbf, const float* __restrict__ Wemb,
                            const float* __restrict__ bemb, float* __restrict__ edge_feat, int E) {
  int idx = blockIdx.x * blockDim.x + threadIdx.x;
  if (idx >= E * DFEAT) return;
  int e = idx >> 6, d = idx & 63;
  float a = bemb[d];
#pragma unroll
  for (int j = 0; j < 9; ++j) a += rbf[(size_t)e * 9 + j] * Wemb[j * DFEAT + d];
  edge_feat[idx] = siluf(a);
}

// ---------------------------------------------------------------------------
// 3) Triple (line-graph) basis precompute: three_basis[T,9], w_tri, end_atom
// ---------------------------------------------------------------------------
__global__ void k_triple_basis(const int* __restrict__ tsrc, const int* __restrict__ tdst,
                               const int* __restrict__ edst, const float* __restrict__ bond_vec,
                               const float* __restrict__ dist, const float* __restrict__ tbcut,
                               float* __restrict__ tb3, float* __restrict__ wtri,
                               int* __restrict__ endat, int T) {
  int t = blockIdx.x * blockDim.x + threadIdx.x;
  if (t >= T) return;
  int ts = tsrc[t], td = tdst[t];
  float ax = bond_vec[ts * 3 + 0], ay = bond_vec[ts * 3 + 1], az = bond_vec[ts * 3 + 2];
  float bx = bond_vec[td * 3 + 0], by = bond_vec[td * 3 + 1], bz = bond_vec[td * 3 + 2];
  float ct = (ax * bx + ay * by + az * bz) / (dist[ts] * dist[td]);
  ct = fminf(fmaxf(ct, -1.0f), 1.0f);
  float sb[9];
  sbf9(dist[td], sb);
  float shf[3];
  shf[0] = 0.28209479177387814f;                                 // Y00
  shf[1] = 0.4886025119029199f * ct;                             // Y10
  shf[2] = 0.6307831305050401f * 0.5f * (3.0f * ct * ct - 1.0f); // Y20
#pragma unroll
  for (int l = 0; l < 3; ++l)
#pragma unroll
    for (int n = 0; n < 3; ++n) tb3[(size_t)t * 9 + l * 3 + n] = sb[l * 3 + n] * shf[l];
  wtri[t] = tbcut[ts] * tbcut[td];
  endat[t] = edst[td];
}

// ---------------------------------------------------------------------------
// 4) Weight prep: f32 [K,64] -> bf16 WMMA B-fragment layout
//    idx = ((nt*NK + kc)*32 + lane)*16 + j ; lane<16: K=kc*32+(j<8?j:j+8), lane>=16: +8
// ---------------------------------------------------------------------------
__global__ void k_wprep(const float* __restrict__ W, bf16_t* __restrict__ out, int K) {
  int idx = blockIdx.x * blockDim.x + threadIdx.x;
  int NK = K / 32;
  int total = 4 * NK * 512;
  if (idx >= total) return;
  int j = idx & 15;
  int lane = (idx >> 4) & 31;
  int blk = idx >> 9;  // nt*NK + kc
  int kc = blk % NK, nt = blk / NK;
  int col = nt * 16 + (lane & 15);
  int k = kc * 32 + ((lane >= 16) ? 8 : 0) + (j & 7) + ((j & 8) ? 16 : 0);
  out[idx] = (bf16_t)W[k * DFEAT + col];
}

// ---------------------------------------------------------------------------
// 5) Per-block three-body path (VALU + atomics; K=9, memory bound)
// ---------------------------------------------------------------------------
__global__ void k_atoms(const float* __restrict__ node_feat, const float* __restrict__ tbWa,
                        const float* __restrict__ tbba, float* __restrict__ atoms, int N) {
  int idx = blockIdx.x * blockDim.x + threadIdx.x;
  if (idx >= N * 9) return;
  int n = idx / 9, j = idx - n * 9;
  float a = tbba[j];
#pragma unroll 16
  for (int k = 0; k < DFEAT; ++k) a += node_feat[(size_t)n * DFEAT + k] * tbWa[k * 9 + j];
  atoms[idx] = sigmf_(a);
}

__global__ void k_triple_scatter(const float* __restrict__ tb3, const float* __restrict__ wtri,
                                 const int* __restrict__ endat, const float* __restrict__ atoms,
                                 const int* __restrict__ tsrc, float* __restrict__ nb, int T) {
  int t = blockIdx.x * blockDim.x + threadIdx.x;
  if (t >= T) return;
  int es = tsrc[t];
  int ea = endat[t];
  float w = wtri[t];
#pragma unroll
  for (int j = 0; j < 9; ++j) {
    float v = tb3[(size_t)t * 9 + j] * atoms[(size_t)ea * 9 + j] * w;
    atomicAdd(&nb[(size_t)es * 9 + j], v);
  }
}

__global__ void k_edge_threebody(const float* __restrict__ nb, const float* __restrict__ tbWl,
                                 const float* __restrict__ tbWg, float* __restrict__ edge_feat,
                                 int E) {
  int idx = blockIdx.x * blockDim.x + threadIdx.x;
  if (idx >= E * DFEAT) return;
  int e = idx >> 6, d = idx & 63;
  float h = 0.0f, g = 0.0f;
#pragma unroll
  for (int j = 0; j < 9; ++j) {
    float v = nb[(size_t)e * 9 + j];
    h += v * tbWl[j * DFEAT + d];
    g += v * tbWg[j * DFEAT + d];
  }
  edge_feat[idx] += siluf(h) * sigmf_(g);
}

// ---------------------------------------------------------------------------
// 6) Gated MLP on WMMA. 8 waves / workgroup, one 16-edge tile per wave.
//    Per-conv weight fragments (81920 B bf16) staged once into LDS, read via
//    ds_load. A layout (16-bit, 16x32): lane<16: row=lane, elems 0..7 ->
//    K=c*32+0..7, 8..15 -> K=c*32+16..23 ; lane>=16: row=lane-16, K +8.
//    C/D layout: vgpr i, lane<16 -> (M=i,N=lane); lane>=16 -> (M=i+8,N=lane-16)
// ---------------------------------------------------------------------------
template <int NK>
__device__ __forceinline__ void gemm_tile(const v16bf* A, const bf16_t* W,
                                          const float* __restrict__ bias, int row, int lane,
                                          v8f* acc) {
#pragma unroll
  for (int nt = 0; nt < 4; ++nt) {
    float bv = bias[nt * 16 + row];
    v8f a;
#pragma unroll
    for (int i = 0; i < 8; ++i) a[i] = bv;
#pragma unroll
    for (int c = 0; c < NK; ++c) {
      v16bf bfrag = *(const v16bf*)(W + ((size_t)(nt * NK + c) * 32 + (size_t)lane) * 16);
      a = __builtin_amdgcn_wmma_f32_16x16x32_bf16(false, A[c], false, bfrag, (short)0, a,
                                                  false, false);
    }
    acc[nt] = a;
  }
}

// MODE 0: edge update (edge_feat += me * (rbf@Wr), in place)
// MODE 1: node message (atomicAdd node_accum[dst] += me * (rbf@Wr))
template <int MODE>
__global__ __launch_bounds__(GWAVES * 32) void k_gated_gemm(
    const float* __restrict__ node_feat, float* __restrict__ edge_feat,
    const float* __restrict__ rbf, const int* __restrict__ esrc, const int* __restrict__ edst,
    const bf16_t* __restrict__ wconv,  // PER_CONV_ELEMS bf16: W0|W1|W2|V0|V1|V2
    const float* __restrict__ bb, const float* __restrict__ cc, const float* __restrict__ wr,
    float* __restrict__ node_accum, int E) {
  extern __shared__ char smem[];
  bf16_t* lw = (bf16_t*)smem;

  // cooperative weight staging: 81920 B = 5120 uint4, 20 per thread
  {
    const uint4* s4 = (const uint4*)wconv;
    uint4* d4 = (uint4*)smem;
#pragma unroll
    for (int i = 0; i < SMEM_W_BYTES / 16 / (GWAVES * 32); ++i)
      d4[i * (GWAVES * 32) + threadIdx.x] = s4[i * (GWAVES * 32) + threadIdx.x];
  }

  const int wave = threadIdx.x >> 5;
  const int lane = threadIdx.x & 31;
  const int row = lane & 15;
  const bool hi = lane >= 16;
  float* srbf = (float*)(smem + SMEM_SRBF_OFF) + wave * 16 * 9;   // [16][9]
  bf16_t* sh  = (bf16_t*)(smem + SMEM_SH_OFF) + wave * 16 * DFEAT; // [16][64]

  const int tile = blockIdx.x * GWAVES + wave;
  int e = tile * 16 + row;
  if (e >= E) e = E - 1;

  if (!hi) {
#pragma unroll
    for (int j = 0; j < 9; ++j) srbf[row * 9 + j] = rbf[(size_t)e * 9 + j];
  }

  const int s = esrc[e], dn = edst[e];

  // Build A0 fragments for xe = [v_i | v_j | edge_feat], 6 chunks of K=32.
  v16bf a0[6];
#pragma unroll
  for (int c = 0; c < 6; ++c) {
#pragma unroll
    for (int g = 0; g < 2; ++g) {
      int koff = c * 32 + g * 16 + (hi ? 8 : 0);
      int region = koff >> 6;
      int col = koff & 63;
      const float* src;
      if (region == 0)      src = node_feat + (size_t)s * DFEAT + col;
      else if (region == 1) src = node_feat + (size_t)dn * DFEAT + col;
      else                  src = edge_feat + (size_t)e * DFEAT + col;
#pragma unroll
      for (int j = 0; j < 8; ++j) a0[c][g * 8 + j] = (bf16_t)src[j];
    }
  }
  __syncthreads();  // weights + srbf visible

  const bf16_t* w0 = lw;
  const bf16_t* w1 = lw + 12288;
  const bf16_t* w2 = lw + 16384;
  const bf16_t* v0 = lw + 20480;
  const bf16_t* v1 = lw + 32768;
  const bf16_t* v2 = lw + 36864;

  v8f acc[4];
  v16bf a1[2];

  auto act_store = [&](int act) {  // activate C-layout acc, restripe into LDS bf16
#pragma unroll
    for (int nt = 0; nt < 4; ++nt)
#pragma unroll
      for (int i = 0; i < 8; ++i) {
        float x = acc[nt][i];
        x = (act == 0) ? siluf(x) : sigmf_(x);
        sh[(i + (hi ? 8 : 0)) * DFEAT + nt * 16 + row] = (bf16_t)x;
      }
  };
  auto loadA2 = [&]() {  // LDS [16][64] bf16 -> A fragments (K=64, 2 chunks)
#pragma unroll
    for (int c = 0; c < 2; ++c)
#pragma unroll
      for (int g = 0; g < 2; ++g) {
        int koff = c * 32 + g * 16 + (hi ? 8 : 0);
#pragma unroll
        for (int j = 0; j < 8; ++j) a1[c][g * 8 + j] = sh[row * DFEAT + koff + j];
      }
  };

  // ---- main path: silu(silu(silu(x@W0)@W1)@W2) ----
  gemm_tile<6>(a0, w0, bb + 0, row, lane, acc);
  act_store(0); __syncthreads();
  loadA2(); __syncthreads();
  gemm_tile<2>(a1, w1, bb + 64, row, lane, acc);
  act_store(0); __syncthreads();
  loadA2(); __syncthreads();
  gemm_tile<2>(a1, w2, bb + 128, row, lane, acc);
  v8f outm[4];
#pragma unroll
  for (int nt = 0; nt < 4; ++nt)
#pragma unroll
    for (int i = 0; i < 8; ++i) outm[nt][i] = siluf(acc[nt][i]);

  // ---- gate path: sigmoid(silu(silu(x@V0)@V1)@V2) ----
  gemm_tile<6>(a0, v0, cc + 0, row, lane, acc);
  act_store(0); __syncthreads();
  loadA2(); __syncthreads();
  gemm_tile<2>(a1, v1, cc + 64, row, lane, acc);
  act_store(0); __syncthreads();
  loadA2(); __syncthreads();
  gemm_tile<2>(a1, v2, cc + 128, row, lane, acc);

  // ---- epilogue: me = outm * sigmoid(acc); scale by (rbf @ Wr) ----
#pragma unroll
  for (int nt = 0; nt < 4; ++nt) {
    int col = nt * 16 + row;
    float wcol[9];
#pragma unroll
    for (int j = 0; j < 9; ++j) wcol[j] = wr[j * DFEAT + col];
#pragma unroll
    for (int i = 0; i < 8; ++i) {
      int m = i + (hi ? 8 : 0);
      int erow = tile * 16 + m;
      if (erow >= E) continue;
      float rw = 0.0f;
#pragma unroll
      for (int j = 0; j < 9; ++j) rw += srbf[m * 9 + j] * wcol[j];
      float val = outm[nt][i] * sigmf_(acc[nt][i]) * rw;
      if (MODE == 0) {
        float* p = edge_feat + (size_t)erow * DFEAT + col;
        *p = *p + val;
      } else {
        int dd = edst[erow];
        atomicAdd(&node_accum[(size_t)dd * DFEAT + col], val);
      }
    }
  }
}

// ---------------------------------------------------------------------------
// 7) node_feat += accumulated messages; pooling; final MLP
// ---------------------------------------------------------------------------
__global__ void k_node_add(float* __restrict__ node_feat, const float* __restrict__ acc, int N) {
  int idx = blockIdx.x * blockDim.x + threadIdx.x;
  if (idx < N * DFEAT) node_feat[idx] += acc[idx];
}

__global__ void k_pool(const float* __restrict__ node_feat, const int* __restrict__ gidx,
                       float* __restrict__ pooled, float* __restrict__ counts, int N) {
  int idx = blockIdx.x * blockDim.x + threadIdx.x;
  if (idx >= N * DFEAT) return;
  int n = idx >> 6, d = idx & 63;
  int g = gidx[n];
  atomicAdd(&pooled[(size_t)g * DFEAT + d], node_feat[idx]);
  if (d == 0) atomicAdd(&counts[g], 1.0f);
}

__global__ void k_final(const float* __restrict__ pooled, const float* __restrict__ counts,
                        const float* __restrict__ fW0, const float* __restrict__ fb0,
                        const float* __restrict__ fW1, const float* __restrict__ fb1,
                        const float* __restrict__ fW2, const float* __restrict__ fb2,
                        float* __restrict__ out) {
  int g = blockIdx.x, d = threadIdx.x;
  __shared__ float x[DFEAT], h[DFEAT];
  float cnt = fmaxf(counts[g], 1.0f);
  x[d] = pooled[(size_t)g * DFEAT + d] / cnt;
  __syncthreads();
  float a = fb0[d];
  for (int k = 0; k < DFEAT; ++k) a += x[k] * fW0[k * DFEAT + d];
  __syncthreads();
  h[d] = siluf(a);
  __syncthreads();
  float b = fb1[d];
  for (int k = 0; k < DFEAT; ++k) b += h[k] * fW1[k * DFEAT + d];
  __syncthreads();
  x[d] = siluf(b);
  __syncthreads();
  if (d == 0) {
    float o = fb2[0];
    for (int k = 0; k < DFEAT; ++k) o += x[k] * fW2[k];
    out[g] = o;
  }
}

// ---------------------------------------------------------------------------
// host-side orchestration
// ---------------------------------------------------------------------------
extern "C" void kernel_launch(void* const* d_in, const int* in_sizes, int n_in, void* d_out,
                              int out_size, void* d_ws, size_t ws_size, hipStream_t stream) {
  const float* positions = (const float*)d_in[0];
  const int* node_type   = (const int*)d_in[1];
  const int* edge_src    = (const int*)d_in[2];
  const int* edge_dst    = (const int*)d_in[3];
  const int* triple_src  = (const int*)d_in[4];
  const int* triple_dst  = (const int*)d_in[5];
  const int* graph_idx   = (const int*)d_in[6];
  const float* emb_table = (const float*)d_in[7];
  const float* Wemb      = (const float*)d_in[8];
  const float* bemb      = (const float*)d_in[9];
  const float* tbWa      = (const float*)d_in[10];
  const float* tbba      = (const float*)d_in[11];
  const float* tbWl      = (const float*)d_in[12];
  const float* tbWg      = (const float*)d_in[13];
  const float* convW0    = (const float*)d_in[14];
  const float* convW1    = (const float*)d_in[15];
  const float* convW2    = (const float*)d_in[16];
  const float* convV0    = (const float*)d_in[17];
  const float* convV1    = (const float*)d_in[18];
  const float* convV2    = (const float*)d_in[19];
  const float* convb     = (const float*)d_in[20];
  const float* convc     = (const float*)d_in[21];
  const float* convWr    = (const float*)d_in[22];
  const float* fW0 = (const float*)d_in[23];
  const float* fb0 = (const float*)d_in[24];
  const float* fW1 = (const float*)d_in[25];
  const float* fb1 = (const float*)d_in[26];
  const float* fW2 = (const float*)d_in[27];
  const float* fb2 = (const float*)d_in[28];

  const int N = in_sizes[1];
  const int E = in_sizes[2];
  const int T = in_sizes[4];

  // workspace carving (256B aligned)
  char* base = (char*)d_ws;
  size_t off = 0;
  auto take = [&](size_t bytes) -> char* {
    char* p = base + off;
    off += (bytes + 255) & ~(size_t)255;
    return p;
  };
  float* bond_vec  = (float*)take((size_t)E * 3 * 4);
  float* dist      = (float*)take((size_t)E * 4);
  float* rbf       = (float*)take((size_t)E * 9 * 4);
  float* tbcut     = (float*)take((size_t)E * 4);
  float* node_feat = (float*)take((size_t)N * DFEAT * 4);
  float* edge_feat = (float*)take((size_t)E * DFEAT * 4);
  float* node_acc  = (float*)take((size_t)N * DFEAT * 4);
  float* nb        = (float*)take((size_t)E * 9 * 4);
  float* atoms     = (float*)take((size_t)N * 9 * 4);
  float* tb3       = (float*)take((size_t)T * 9 * 4);
  float* wtri      = (float*)take((size_t)T * 4);
  int*   endat     = (int*)take((size_t)T * 4);
  float* pooled    = (float*)take((size_t)NGRAPH * DFEAT * 4);
  float* counts    = (float*)take((size_t)NGRAPH * 4);
  bf16_t* wprep    = (bf16_t*)take((size_t)PER_CONV_ELEMS * 6 * sizeof(bf16_t));
  (void)ws_size; (void)n_in; (void)out_size;

  const int TPB = 256;
  // --- static precompute ---
  k_edge_geom<<<(E + TPB - 1) / TPB, TPB, 0, stream>>>(positions, edge_src, edge_dst, bond_vec,
                                                       dist, rbf, tbcut, E);
  k_node_init<<<((size_t)N * DFEAT + TPB - 1) / TPB, TPB, 0, stream>>>(emb_table, node_type,
                                                                       node_feat, N);
  k_edge_init<<<((size_t)E * DFEAT + TPB - 1) / TPB, TPB, 0, stream>>>(rbf, Wemb, bemb, edge_feat,
                                                                       E);
  k_triple_basis<<<(T + TPB - 1) / TPB, TPB, 0, stream>>>(triple_src, triple_dst, edge_dst,
                                                          bond_vec, dist, tbcut, tb3, wtri, endat,
                                                          T);
  // weight prep (bf16 WMMA fragments)
  for (int b = 0; b < NBLK; ++b)
    for (int cv = 0; cv < 2; ++cv) {
      int bc = b * 2 + cv;
      bf16_t* wp = wprep + (size_t)bc * PER_CONV_ELEMS;
      k_wprep<<<(12288 + TPB - 1) / TPB, TPB, 0, stream>>>(convW0 + (size_t)bc * 192 * 64, wp, 192);
      k_wprep<<<(4096 + TPB - 1) / TPB, TPB, 0, stream>>>(convW1 + (size_t)bc * 64 * 64, wp + 12288, 64);
      k_wprep<<<(4096 + TPB - 1) / TPB, TPB, 0, stream>>>(convW2 + (size_t)bc * 64 * 64, wp + 16384, 64);
      k_wprep<<<(12288 + TPB - 1) / TPB, TPB, 0, stream>>>(convV0 + (size_t)bc * 192 * 64, wp + 20480, 192);
      k_wprep<<<(4096 + TPB - 1) / TPB, TPB, 0, stream>>>(convV1 + (size_t)bc * 64 * 64, wp + 32768, 64);
      k_wprep<<<(4096 + TPB - 1) / TPB, TPB, 0, stream>>>(convV2 + (size_t)bc * 64 * 64, wp + 36864, 64);
    }

  const int ETILES = (E + 15) / 16;
  const int GGRID = (ETILES + GWAVES - 1) / GWAVES;
  // --- message passing blocks ---
  for (int b = 0; b < NBLK; ++b) {
    k_atoms<<<((size_t)N * 9 + TPB - 1) / TPB, TPB, 0, stream>>>(
        node_feat, tbWa + (size_t)b * 64 * 9, tbba + (size_t)b * 9, atoms, N);
    hipMemsetAsync(nb, 0, (size_t)E * 9 * 4, stream);
    k_triple_scatter<<<(T + TPB - 1) / TPB, TPB, 0, stream>>>(tb3, wtri, endat, atoms, triple_src,
                                                              nb, T);
    k_edge_threebody<<<((size_t)E * DFEAT + TPB - 1) / TPB, TPB, 0, stream>>>(
        nb, tbWl + (size_t)b * 9 * 64, tbWg + (size_t)b * 9 * 64, edge_feat, E);

    // edge update (conv 0)
    {
      int bc = b * 2 + 0;
      k_gated_gemm<0><<<GGRID, GWAVES * 32, SMEM_TOTAL, stream>>>(
          node_feat, edge_feat, rbf, edge_src, edge_dst, wprep + (size_t)bc * PER_CONV_ELEMS,
          convb + (size_t)bc * 3 * 64, convc + (size_t)bc * 3 * 64,
          convWr + (size_t)bc * 9 * 64, node_acc, E);
    }
    // node message (conv 1)
    hipMemsetAsync(node_acc, 0, (size_t)N * DFEAT * 4, stream);
    {
      int bc = b * 2 + 1;
      k_gated_gemm<1><<<GGRID, GWAVES * 32, SMEM_TOTAL, stream>>>(
          node_feat, edge_feat, rbf, edge_src, edge_dst, wprep + (size_t)bc * PER_CONV_ELEMS,
          convb + (size_t)bc * 3 * 64, convc + (size_t)bc * 3 * 64,
          convWr + (size_t)bc * 9 * 64, node_acc, E);
    }
    k_node_add<<<((size_t)N * DFEAT + TPB - 1) / TPB, TPB, 0, stream>>>(node_feat, node_acc, N);
  }

  // --- readout ---
  hipMemsetAsync(pooled, 0, (size_t)NGRAPH * DFEAT * 4, stream);
  hipMemsetAsync(counts, 0, (size_t)NGRAPH * 4, stream);
  k_pool<<<((size_t)N * DFEAT + TPB - 1) / TPB, TPB, 0, stream>>>(node_feat, graph_idx, pooled,
                                                                  counts, N);
  k_final<<<NGRAPH, DFEAT, 0, stream>>>(pooled, counts, fW0, fb0, fW1, fb1, fW2, fb2,
                                        (float*)d_out);
}
